// LatentEdgeScorer_28905129902363
// MI455X (gfx1250) — matrix operand order repaired and checked by reading, not compiled, not run on hardware
//
#include <hip/hip_runtime.h>
#include <hip/hip_bf16.h>
#include <math.h>

#define NODE_DIM 256
#define EDGE_H   128
#define BB       2
#define NN       512

typedef __bf16 v16bf __attribute__((ext_vector_type(16)));
typedef __bf16 v8bf  __attribute__((ext_vector_type(8)));
typedef float  v8f   __attribute__((ext_vector_type(8)));

union Frag { v16bf v; v8bf h[2]; };

__device__ __forceinline__ unsigned short f32_to_bf16_rne(float f) {
    unsigned int u = __float_as_uint(f);
    u += 0x7FFFu + ((u >> 16) & 1u);   // round-to-nearest-even
    return (unsigned short)(u >> 16);
}

// ---------------------------------------------------------------------------
// Phase 1: per row (b,n): shared-moment layernorm (two affine variants),
// src = LNs(x)@sW^T + sb ; tgt = LNt(x)@tW^T + tb ; srcWs = src@Ws^T
// (the tgt@Wt^T term is folded into phase 2's B matrix, so no tgtWt output)
// Outputs: src (f32), tgt (bf16 -> WMMA A-frags), srcWs (f32)
// ---------------------------------------------------------------------------
__global__ __launch_bounds__(128)
void phase1_norm_proj(const float* __restrict__ latent,
                      const float* __restrict__ sn_g, const float* __restrict__ sn_b,
                      const float* __restrict__ tn_g, const float* __restrict__ tn_b,
                      const float* __restrict__ sW,   const float* __restrict__ sb,
                      const float* __restrict__ tW,   const float* __restrict__ tb,
                      const float* __restrict__ W1,
                      float* __restrict__ src_f32,
                      unsigned short* __restrict__ tgt_bf16,
                      float* __restrict__ srcWs)
{
    __shared__ float xs[NODE_DIM];
    __shared__ float xt[NODE_DIM];
    __shared__ float s_sh[EDGE_H];
    __shared__ float red[8];
    __shared__ float stats[2];

    const int row = blockIdx.x;               // b*N + n
    const int tid = threadIdx.x;              // 0..127
    const float* x = latent + (size_t)row * NODE_DIM;

    float x0 = x[tid];
    float x1 = x[tid + 128];
    float s  = x0 + x1;
    float s2 = x0 * x0 + x1 * x1;
    #pragma unroll
    for (int m = 16; m >= 1; m >>= 1) {       // wave32 reduction
        s  += __shfl_xor(s,  m, 32);
        s2 += __shfl_xor(s2, m, 32);
    }
    const int wave = tid >> 5, lane = tid & 31;
    if (lane == 0) { red[wave] = s; red[wave + 4] = s2; }
    __syncthreads();
    if (tid == 0) {
        float ts  = red[0] + red[1] + red[2] + red[3];
        float tq  = red[4] + red[5] + red[6] + red[7];
        float mu  = ts * (1.0f / NODE_DIM);
        float var = tq * (1.0f / NODE_DIM) - mu * mu;
        stats[0] = mu;
        stats[1] = rsqrtf(var + 1e-5f);
    }
    __syncthreads();
    const float mu = stats[0], rs = stats[1];
    {
        float h0 = (x0 - mu) * rs;
        float h1 = (x1 - mu) * rs;
        xs[tid]       = h0 * sn_g[tid]       + sn_b[tid];
        xs[tid + 128] = h1 * sn_g[tid + 128] + sn_b[tid + 128];
        xt[tid]       = h0 * tn_g[tid]       + tn_b[tid];
        xt[tid + 128] = h1 * tn_g[tid + 128] + tn_b[tid + 128];
    }
    __syncthreads();

    {   // src[k], tgt[k]
        const int k = tid;
        const float* wsrow = sW + (size_t)k * NODE_DIM;
        const float* wtrow = tW + (size_t)k * NODE_DIM;
        float a = sb[k], t = tb[k];
        #pragma unroll 4
        for (int h = 0; h < NODE_DIM; ++h) {
            a = fmaf(xs[h], wsrow[h], a);
            t = fmaf(xt[h], wtrow[h], t);
        }
        s_sh[k] = a;
        src_f32 [(size_t)row * EDGE_H + k] = a;
        tgt_bf16[(size_t)row * EDGE_H + k] = f32_to_bf16_rne(t);
    }
    __syncthreads();

    {   // srcWs[k] = src . W1[k, 0:128]
        const int k = tid;
        const float* wr = W1 + (size_t)k * (3 * EDGE_H);
        float a = 0.f;
        #pragma unroll 4
        for (int h = 0; h < EDGE_H; ++h) a = fmaf(s_sh[h], wr[h], a);
        srcWs[(size_t)row * EDGE_H + k] = a;
    }
}

// ---------------------------------------------------------------------------
// Phase 2: one workgroup per (b,i).
//   B matrix (LDS, bf16):  W''[k,h] = Wt[k,h] + Wp[k,h]*src_i[h]
//     => tgt @ W''^T  =  tgt@Wt^T  +  (tgt (.) src_i) @ Wp^T   (one GEMM
//        covers both the broadcast-over-i term and the trilinear term)
//   acc[j,k] via v_wmma_f32_16x16x32_bf16 (K=128 -> 4 chained wmma)
//   val = acc + (srcWs_i[k]+b1[k]);  g = gelu_exact(val)
//   logits[j] = sum_k g*W2[k] + b2 ; edges = softsign(logits); diag = 0
// 8 waves; each wave owns a 16-row j-tile x all 8 k-tiles (8x v8f accum).
// ---------------------------------------------------------------------------
__global__ __launch_bounds__(256)
void phase2_edges(const float* __restrict__ W1,
                  const float* __restrict__ b1,
                  const float* __restrict__ W2,
                  const float* __restrict__ b2,
                  const float* __restrict__ src_f32,
                  const unsigned short* __restrict__ tgt_bf16,
                  const float* __restrict__ srcWs,
                  float* __restrict__ out)
{
    __shared__ __align__(16) unsigned short lds_w[EDGE_H * EDGE_H]; // 32 KB
    __shared__ float ls_c [EDGE_H];   // srcWs_i[k] + b1[k]
    __shared__ float ls_w2[EDGE_H];

    const int blk = blockIdx.x;
    const int b   = blk >> 9;          // N = 512
    const int i   = blk & (NN - 1);
    const int tid = threadIdx.x;

    const float* src_i = src_f32 + (size_t)(b * NN + i) * EDGE_H;

    if (tid < EDGE_H) {
        ls_c [tid] = srcWs[(size_t)(b * NN + i) * EDGE_H + tid] + b1[tid];
        ls_w2[tid] = W2[tid];
    }
    for (int idx = tid; idx < EDGE_H * EDGE_H; idx += 256) {
        const int k = idx >> 7, h = idx & 127;
        const float* w1row = W1 + (size_t)k * 384;
        // W'' = Wt + Wp (.) src_i   (Wt = W1[:,128:256], Wp = W1[:,256:384])
        const float v = fmaf(w1row[256 + h], src_i[h], w1row[128 + h]);
        lds_w[idx] = f32_to_bf16_rne(v);
    }
    __syncthreads();

    const int wave = tid >> 5;
    const int lane = tid & 31;
    const int half = lane >> 4;        // ISA frag layout: lane halves split K
    const int l16  = lane & 15;
    const float b2v = b2[0];

    for (int jb = 0; jb < 4; ++jb) {
        const int j0 = jb * 128 + wave * 16;

        // A fragments: 16 tgt rows (bf16, row-major), K = 0..127 in 4 chunks.
        // Lane L holds row (L&15); chunk c: cols c*32 + (L>>4)*8 + {0..7, 16..23}.
        Frag aa[4];
        const unsigned short* arow =
            tgt_bf16 + (size_t)(b * NN + j0 + l16) * EDGE_H + half * 8;
        #pragma unroll
        for (int c = 0; c < 4; ++c) {
            aa[c].h[0] = *(const v8bf*)(arow + c * 32);
            aa[c].h[1] = *(const v8bf*)(arow + c * 32 + 16);
        }

        v8f acc[8];
        #pragma unroll
        for (int kt = 0; kt < 8; ++kt)
            acc[kt] = (v8f){0.f,0.f,0.f,0.f,0.f,0.f,0.f,0.f};

        // B fragments from LDS: lane L holds W'' row (k-tile col = L&15),
        // K per lane = c*32 + (L>>4)*8 + {0..7, 16..23} (2x b128).
        #pragma unroll
        for (int c = 0; c < 4; ++c) {
            #pragma unroll
            for (int kt = 0; kt < 8; ++kt) {
                Frag bb;
                const unsigned short* bp =
                    &lds_w[(kt * 16 + l16) * EDGE_H + c * 32 + half * 8];
                bb.h[0] = *(const v8bf*)(bp);
                bb.h[1] = *(const v8bf*)(bp + 16);
                acc[kt] = __builtin_amdgcn_wmma_f32_16x16x32_bf16(
                    false, aa[c].v, false, bb.v, (short)0, acc[kt], false, false);
            }
        }

        // Epilogue (pure VALU now).
        // C layout: VGPR v, lane L -> (j = j0 + v + 8*(L>>4), k = kt*16 + (L&15))
        float partial[8] = {0.f,0.f,0.f,0.f,0.f,0.f,0.f,0.f};
        #pragma unroll
        for (int kt = 0; kt < 8; ++kt) {
            const int k    = kt * 16 + l16;
            const float ck  = ls_c[k];
            const float w2k = ls_w2[k];
            #pragma unroll
            for (int v = 0; v < 8; ++v) {
                const float val = acc[kt][v] + ck;
                const float g = 0.5f * val * (1.0f + erff(val * 0.70710678118654752f));
                partial[v] = fmaf(g, w2k, partial[v]);
            }
        }
        // k-reduction across the 16 lanes of each half (same k set per half).
        #pragma unroll
        for (int v = 0; v < 8; ++v) {
            float p = partial[v];
            p += __shfl_xor(p, 1, 16);
            p += __shfl_xor(p, 2, 16);
            p += __shfl_xor(p, 4, 16);
            p += __shfl_xor(p, 8, 16);
            if (l16 == 0) {
                const int j = j0 + v + (half << 3);
                const float l = p + b2v;
                float e = l / (1.0f + fabsf(l));
                if (j == i) e = 0.0f;
                out[((size_t)b * NN + i) * NN + j] = e;
            }
        }
    }
}

extern "C" void kernel_launch(void* const* d_in, const int* in_sizes, int n_in,
                              void* d_out, int out_size, void* d_ws, size_t ws_size,
                              hipStream_t stream) {
    const float* latent = (const float*)d_in[0];
    const float* sn_g   = (const float*)d_in[1];
    const float* sn_b   = (const float*)d_in[2];
    const float* tn_g   = (const float*)d_in[3];
    const float* tn_b   = (const float*)d_in[4];
    const float* sW     = (const float*)d_in[5];
    const float* sb     = (const float*)d_in[6];
    const float* tW     = (const float*)d_in[7];
    const float* tb     = (const float*)d_in[8];
    const float* W1     = (const float*)d_in[9];
    const float* b1     = (const float*)d_in[10];
    const float* W2     = (const float*)d_in[11];
    const float* b2     = (const float*)d_in[12];
    float* out = (float*)d_out;

    // Workspace carve (all 16B aligned):
    char* ws = (char*)d_ws;
    float*          src_f32  = (float*)(ws);                    // 512 KB
    unsigned short* tgt_bf16 = (unsigned short*)(ws + 524288);  // 256 KB
    float*          srcWs    = (float*)(ws + 524288 + 262144);  // 512 KB

    const int rows = BB * NN;  // 1024
    phase1_norm_proj<<<dim3(rows), dim3(128), 0, stream>>>(
        latent, sn_g, sn_b, tn_g, tn_b, sW, sb, tW, tb, W1,
        src_f32, tgt_bf16, srcWs);

    phase2_edges<<<dim3(rows), dim3(256), 0, stream>>>(
        W1, b1, W2, b2, src_f32, tgt_bf16, srcWs, out);
}
